// MambaLayer_25159918419974
// MI455X (gfx1250) — compile-verified, matrix operand order checked by reading, not run on hardware
//
#include <hip/hip_runtime.h>
#include <hip/hip_bf16.h>
#include <math.h>

// ---------------------------------------------------------------------------
// Bidirectional Mamba block for MI455X (gfx1250, wave32, WMMA).
// All GEMMs use v_wmma_f32_16x16x32_f16 (f16 inputs, f32 accumulate).
// Token-major activation layout so A-fragments and (out,in)-row-major weights
// both load as two contiguous 16B chunks per lane (matches CDNA5 fragment
// layout: lane<16 -> K {0..7,16..23}, lane>=16 -> K {8..15,24..31}).
// GEMM is templated on <EPI, MT, NT>: each wave owns a (16*MT) x (16*NT)
// output block, so the j-loop/mi-loop fully unroll into straight-line WMMA
// (no exec-mask predication; all problem dims divide the strip exactly).
// ---------------------------------------------------------------------------

typedef __attribute__((ext_vector_type(16))) _Float16 v16h;
typedef __attribute__((ext_vector_type(8)))  _Float16 v8h;
typedef __attribute__((ext_vector_type(8)))  float    v8f;

union F16Frag { v16h v; v8h h[2]; };

#define BSZ     4
#define SEQL    1024
#define CDIM    256
#define DINNER  512
#define NSTATE  16
#define DTRANK  16
#define KXP     48      // dt_rank + 2*N
#define XDBL_LD 64      // padded row stride for x_dbl

// epilogue modes
#define EPI_NONE          0
#define EPI_SOFTPLUS_BIAS 1
#define EPI_ADD           2
#define EPI_STORE_F16     3

// ---------------------------------------------------------------------------
// LN1: read x (B, C, L), produce residual copy x_flat (B*L, C) f32 and
// normalized xn16 (B*L, C) f16 for the in_proj WMMA GEMM.
// ---------------------------------------------------------------------------
__global__ void ln1_kernel(const float* __restrict__ x,
                           const float* __restrict__ w,
                           const float* __restrict__ b,
                           float* __restrict__ xflat,
                           _Float16* __restrict__ xn16)
{
    const int token = blockIdx.x;              // b*L + l
    const int bb = token / SEQL;
    const int l  = token - bb * SEQL;
    const int c  = threadIdx.x;                // 0..255
    __shared__ float red[CDIM];

    float v = x[((size_t)bb * CDIM + c) * SEQL + l];
    red[c] = v;
    __syncthreads();
    for (int s = CDIM / 2; s > 0; s >>= 1) {
        if (c < s) red[c] += red[c + s];
        __syncthreads();
    }
    const float mu = red[0] * (1.0f / CDIM);
    __syncthreads();
    const float xc = v - mu;
    red[c] = xc * xc;
    __syncthreads();
    for (int s = CDIM / 2; s > 0; s >>= 1) {
        if (c < s) red[c] += red[c + s];
        __syncthreads();
    }
    const float var = red[0] * (1.0f / CDIM);
    const float nrm = xc * rsqrtf(var + 1e-5f) * w[c] + b[c];

    xflat[(size_t)token * CDIM + c] = v;
    xn16 [(size_t)token * CDIM + c] = (_Float16)nrm;
}

// ---------------------------------------------------------------------------
// LN2 + transpose back to (B, C, L) output.
// ---------------------------------------------------------------------------
__global__ void ln2_out_kernel(const float* __restrict__ ym,
                               const float* __restrict__ w,
                               const float* __restrict__ b,
                               float* __restrict__ out)
{
    const int token = blockIdx.x;
    const int bb = token / SEQL;
    const int l  = token - bb * SEQL;
    const int c  = threadIdx.x;
    __shared__ float red[CDIM];

    float v = ym[(size_t)token * CDIM + c];
    red[c] = v;
    __syncthreads();
    for (int s = CDIM / 2; s > 0; s >>= 1) {
        if (c < s) red[c] += red[c + s];
        __syncthreads();
    }
    const float mu = red[0] * (1.0f / CDIM);
    __syncthreads();
    const float xc = v - mu;
    red[c] = xc * xc;
    __syncthreads();
    for (int s = CDIM / 2; s > 0; s >>= 1) {
        if (c < s) red[c] += red[c + s];
        __syncthreads();
    }
    const float var = red[0] * (1.0f / CDIM);
    out[((size_t)bb * CDIM + c) * SEQL + l] =
        xc * rsqrtf(var + 1e-5f) * w[c] + b[c];
}

// ---------------------------------------------------------------------------
// Generic f32 -> f16 convert.
// ---------------------------------------------------------------------------
__global__ void cvt_f32_f16(const float* __restrict__ a,
                            _Float16* __restrict__ o, int n)
{
    int i = blockIdx.x * blockDim.x + threadIdx.x;
    if (i < n) o[i] = (_Float16)a[i];
}

// dtproj_w (D_INNER, 16) -> f16 padded to (D_INNER, 32) with zero cols 16..31
// so the dt GEMM runs a single 16x16x32 WMMA K-step (padded A region is
// multiplied by zero weights, so its contents are irrelevant).
__global__ void cvt_pad_dtproj(const float* __restrict__ w,
                               _Float16* __restrict__ o)
{
    int idx = blockIdx.x * blockDim.x + threadIdx.x;   // DINNER*32
    if (idx >= DINNER * 32) return;
    int k = idx & 31;
    int n = idx >> 5;
    o[idx] = (k < DTRANK) ? (_Float16)w[n * DTRANK + k] : (_Float16)0.0f;
}

// ---------------------------------------------------------------------------
// WMMA GEMM:  C(M,N) = A(M,K) * W(N,K)^T, f16 in / f32 accumulate.
// One wave owns a (16*MT) x (16*NT) output block; K stepped by 32.
// Requires M % (16*MT) == 0 and N % (16*NT) == 0 (true for all call sites),
// so the tile loops fully unroll with no predication around WMMA.
// ---------------------------------------------------------------------------
template <int EPI, int MT, int NT>
__global__ void gemm_wmma_f16(const _Float16* __restrict__ A, int lda,
                              const _Float16* __restrict__ W, int ldw,
                              float* __restrict__ C, int ldc,
                              int M, int N, int K,
                              const float* __restrict__ bias,
                              const float* __restrict__ addsrc,
                              _Float16* __restrict__ C16, int ldc16)
{
    const int lane = threadIdx.x & 31;
    const int wave = threadIdx.x >> 5;
    const int wpb  = blockDim.x >> 5;
    const int l16  = lane & 15;
    const int hi   = lane >> 4;

    const int mStrips = M / (16 * MT);
    const int nStrips = N / (16 * NT);
    const int tasks   = mStrips * nStrips;

    for (int task = blockIdx.x * wpb + wave; task < tasks;
         task += gridDim.x * wpb) {
        const int ms = task % mStrips;
        const int ns = task / mStrips;
        const int m0 = ms * (16 * MT);
        const int n0 = ns * (16 * NT);

        v8f acc[MT][NT];
        #pragma unroll
        for (int mi = 0; mi < MT; ++mi)
            #pragma unroll
            for (int j = 0; j < NT; ++j)
                #pragma unroll
                for (int e = 0; e < 8; ++e) acc[mi][j][e] = 0.0f;

        const _Float16* Arow[MT];
        #pragma unroll
        for (int mi = 0; mi < MT; ++mi)
            Arow[mi] = A + (size_t)(m0 + mi * 16 + l16) * lda;
        const _Float16* Wrow[NT];
        #pragma unroll
        for (int j = 0; j < NT; ++j)
            Wrow[j] = W + (size_t)(n0 + j * 16 + l16) * ldw;

        for (int k0 = 0; k0 < K; k0 += 32) {
            F16Frag af[MT];
            #pragma unroll
            for (int mi = 0; mi < MT; ++mi) {
                af[mi].h[0] = *(const v8h*)(Arow[mi] + k0 + hi * 8);
                af[mi].h[1] = *(const v8h*)(Arow[mi] + k0 + 16 + hi * 8);
                if (k0 + 64 < K)                   // global_prefetch_b8
                    __builtin_prefetch(Arow[mi] + k0 + 64, 0, 1);
            }
            #pragma unroll
            for (int j = 0; j < NT; ++j) {
                F16Frag bf;
                bf.h[0] = *(const v8h*)(Wrow[j] + k0 + hi * 8);
                bf.h[1] = *(const v8h*)(Wrow[j] + k0 + 16 + hi * 8);
                #pragma unroll
                for (int mi = 0; mi < MT; ++mi)
                    acc[mi][j] = __builtin_amdgcn_wmma_f32_16x16x32_f16(
                        false, af[mi].v, false, bf.v, (short)0,
                        acc[mi][j], false, false);
            }
        }

        #pragma unroll
        for (int mi = 0; mi < MT; ++mi) {
            #pragma unroll
            for (int j = 0; j < NT; ++j) {
                const int col = n0 + j * 16 + l16;
                #pragma unroll
                for (int i = 0; i < 8; ++i) {
                    const int row = m0 + mi * 16 + hi * 8 + i;
                    float v = acc[mi][j][i];
                    if (EPI == EPI_SOFTPLUS_BIAS) {
                        v += bias[col];
                        v = (v > 20.0f) ? v : log1pf(__expf(v));
                    }
                    if (EPI == EPI_ADD)
                        v += addsrc[(size_t)row * ldc + col];
                    C[(size_t)row * ldc + col] = v;
                    if (EPI == EPI_STORE_F16)
                        C16[(size_t)row * ldc16 + col] = (_Float16)v;
                }
            }
        }
    }
}

// ---------------------------------------------------------------------------
// Depthwise conv (d_conv=4) + SiLU. dir==0: causal; dir==1: anti-causal
// (equivalent to conv on the reversed sequence, kept in token order).
// xz is (B, L, 2*DINNER) token-major; channel d in [0, DINNER).
// ---------------------------------------------------------------------------
__global__ void conv_silu_kernel(const float* __restrict__ xz,
                                 const float* __restrict__ cw,
                                 const float* __restrict__ cb,
                                 float* __restrict__ xc,
                                 _Float16* __restrict__ xc16, int dir)
{
    const size_t idx = (size_t)blockIdx.x * blockDim.x + threadIdx.x;
    if (idx >= (size_t)BSZ * SEQL * DINNER) return;
    const int d = idx % DINNER;
    const size_t tok = idx / DINNER;
    const int l  = tok % SEQL;
    const int bb = tok / SEQL;

    const float* base = xz + ((size_t)bb * SEQL) * (2 * DINNER) + d;
    float acc = cb[d];
    #pragma unroll
    for (int j = 0; j < 4; ++j) {
        const int lp = (dir == 0) ? (l - 3 + j) : (l + 3 - j);
        if (lp >= 0 && lp < SEQL)
            acc += cw[d * 4 + j] * base[(size_t)lp * (2 * DINNER)];
    }
    const float s = acc / (1.0f + __expf(-acc));
    xc[idx]   = s;
    xc16[idx] = (_Float16)s;
}

// ---------------------------------------------------------------------------
// Selective scan: lane = channel, 16-state recurrence in registers.
// B_t / C_t (32 floats per token) broadcast through LDS each step.
// dir==1 iterates tokens in reverse (backward direction, token order kept).
// Output gated by silu(z) and D*u skip; accumulates into yacc on 2nd pass.
// ---------------------------------------------------------------------------
__global__ void scan_kernel(const float* __restrict__ dt,
                            const float* __restrict__ xdbl,
                            const float* __restrict__ xc,
                            const float* __restrict__ xz,
                            const float* __restrict__ A_log,
                            const float* __restrict__ Dp,
                            float* __restrict__ yacc,
                            int dir, int accumulate)
{
    const int halves = DINNER / 256;                     // 2
    const int bb = blockIdx.x / halves;
    const int d  = (blockIdx.x % halves) * 256 + threadIdx.x;

    float h[NSTATE], Ad[NSTATE];
    #pragma unroll
    for (int n = 0; n < NSTATE; ++n) {
        h[n]  = 0.0f;
        Ad[n] = -__expf(A_log[d * NSTATE + n]);
    }
    const float Dv = Dp[d];
    __shared__ float sBC[2 * NSTATE];

    for (int step = 0; step < SEQL; ++step) {
        const int t = (dir == 0) ? step : (SEQL - 1 - step);
        const size_t tok = (size_t)bb * SEQL + t;
        __syncthreads();
        if (threadIdx.x < 2 * NSTATE)
            sBC[threadIdx.x] = xdbl[tok * XDBL_LD + DTRANK + threadIdx.x];
        __syncthreads();

        const float dtv = dt[tok * DINNER + d];
        const float u   = xc[tok * DINNER + d];
        const float du  = dtv * u;
        float y = 0.0f;
        #pragma unroll
        for (int n = 0; n < NSTATE; ++n) {
            h[n] = __expf(dtv * Ad[n]) * h[n] + du * sBC[n];
            y   += h[n] * sBC[NSTATE + n];
        }
        y += Dv * u;
        const float zv = xz[tok * (2 * DINNER) + DINNER + d];
        y *= zv / (1.0f + __expf(-zv));

        float* o = yacc + tok * DINNER + d;
        if (accumulate) *o += y; else *o = y;
    }
}

// ---------------------------------------------------------------------------
// Host orchestration.
// ---------------------------------------------------------------------------
extern "C" void kernel_launch(void* const* d_in, const int* in_sizes, int n_in,
                              void* d_out, int out_size, void* d_ws, size_t ws_size,
                              hipStream_t stream)
{
    (void)in_sizes; (void)n_in; (void)out_size; (void)ws_size;

    const float* x        = (const float*)d_in[0];
    const float* ln1w     = (const float*)d_in[1];
    const float* ln1b     = (const float*)d_in[2];
    const float* ln2w     = (const float*)d_in[3];
    const float* ln2b     = (const float*)d_in[4];
    const float* inpw     = (const float*)d_in[5];
    const float* convw[2]   = { (const float*)d_in[6],  (const float*)d_in[13] };
    const float* convb[2]   = { (const float*)d_in[7],  (const float*)d_in[14] };
    const float* xprojw[2]  = { (const float*)d_in[8],  (const float*)d_in[15] };
    const float* dtprojw[2] = { (const float*)d_in[9],  (const float*)d_in[16] };
    const float* dtprojb[2] = { (const float*)d_in[10], (const float*)d_in[17] };
    const float* Alog[2]    = { (const float*)d_in[11], (const float*)d_in[18] };
    const float* Dvec[2]    = { (const float*)d_in[12], (const float*)d_in[19] };
    const float* outpw    = (const float*)d_in[20];
    float* out            = (float*)d_out;

    const int TOK = BSZ * SEQL;                  // 4096 tokens

    // -- workspace carve-out (256B aligned) --
    char* ws = (char*)d_ws;
    size_t off = 0;
    auto carve = [&](size_t bytes) {
        char* p = ws + off;
        off = (off + bytes + 255) & ~(size_t)255;
        return p;
    };
    float*    xflat  = (float*)   carve((size_t)TOK * CDIM   * 4);
    _Float16* xn16   = (_Float16*)carve((size_t)TOK * CDIM   * 2);
    _Float16* wi16   = (_Float16*)carve((size_t)2 * DINNER * CDIM * 2);
    _Float16* wx16   = (_Float16*)carve((size_t)KXP * DINNER * 2);
    _Float16* wdt16  = (_Float16*)carve((size_t)DINNER * 32  * 2);
    _Float16* wo16   = (_Float16*)carve((size_t)CDIM * DINNER * 2);
    float*    xzf    = (float*)   carve((size_t)TOK * 2 * DINNER * 4);
    float*    xcf    = (float*)   carve((size_t)TOK * DINNER * 4);
    _Float16* xc16   = (_Float16*)carve((size_t)TOK * DINNER * 2);
    float*    xdblf  = (float*)   carve((size_t)TOK * XDBL_LD * 4);
    _Float16* xdbl16 = (_Float16*)carve((size_t)TOK * XDBL_LD * 2);
    float*    dtf    = (float*)   carve((size_t)TOK * DINNER * 4);
    float*    yaccf  = (float*)   carve((size_t)TOK * DINNER * 4);
    _Float16* yacc16 = (_Float16*)carve((size_t)TOK * DINNER * 2);
    float*    ymf    = (float*)   carve((size_t)TOK * CDIM   * 4);

    // 1) LN1 + transpose + f16 copy
    ln1_kernel<<<TOK, CDIM, 0, stream>>>(x, ln1w, ln1b, xflat, xn16);

    // 2) static weight converts
    {
        int n = 2 * DINNER * CDIM;               // in_proj
        cvt_f32_f16<<<(n + 255) / 256, 256, 0, stream>>>(inpw, wi16, n);
        n = CDIM * DINNER;                       // out_proj
        cvt_f32_f16<<<(n + 255) / 256, 256, 0, stream>>>(outpw, wo16, n);
    }

    // 3) in_proj GEMM: (4096,256) x (1024,256)^T -> xz (B,L,1024)
    //    tasks = (4096/32)*(1024/64) = 2048 waves
    gemm_wmma_f16<EPI_NONE, 2, 4><<<256, 256, 0, stream>>>(
        xn16, CDIM, wi16, CDIM, xzf, 2 * DINNER,
        TOK, 2 * DINNER, CDIM, nullptr, nullptr, nullptr, 0);

    // 4) per-direction pipeline (fwd = 0, bwd = 1)
    for (int dir = 0; dir < 2; ++dir) {
        int n = KXP * DINNER;
        cvt_f32_f16<<<(n + 255) / 256, 256, 0, stream>>>(xprojw[dir], wx16, n);
        cvt_pad_dtproj<<<(DINNER * 32 + 255) / 256, 256, 0, stream>>>(
            dtprojw[dir], wdt16);

        // depthwise conv + silu
        conv_silu_kernel<<<(TOK * DINNER) / 256, 256, 0, stream>>>(
            xzf, convw[dir], convb[dir], xcf, xc16, dir);

        // x_dbl GEMM: (4096,512) x (48,512)^T -> (B,L,48) pad-stride 64 (+f16)
        //    tasks = (4096/32)*(48/48) = 128 waves
        gemm_wmma_f16<EPI_STORE_F16, 2, 3><<<16, 256, 0, stream>>>(
            xc16, DINNER, wx16, DINNER, xdblf, XDBL_LD,
            TOK, KXP, DINNER, nullptr, nullptr, xdbl16, XDBL_LD);

        // dt GEMM: (4096,32 padded) x (512,32)^T -> softplus(v + bias)
        //    tasks = (4096/32)*(512/64) = 1024 waves
        gemm_wmma_f16<EPI_SOFTPLUS_BIAS, 2, 4><<<128, 256, 0, stream>>>(
            xdbl16, XDBL_LD, wdt16, 32, dtf, DINNER,
            TOK, DINNER, 32, dtprojb[dir], nullptr, nullptr, 0);

        // selective scan (reverse order for dir==1), gated, accumulated
        scan_kernel<<<BSZ * (DINNER / 256), 256, 0, stream>>>(
            dtf, xdblf, xcf, xzf, Alog[dir], Dvec[dir], yaccf,
            dir, /*accumulate=*/dir);
    }

    // 5) y -> f16 for out_proj
    {
        int n = TOK * DINNER;
        cvt_f32_f16<<<(n + 255) / 256, 256, 0, stream>>>(yaccf, yacc16, n);
    }

    // 6) out_proj GEMM + residual add: (4096,512) x (256,512)^T + x_flat
    //    tasks = (4096/32)*(256/64) = 512 waves
    gemm_wmma_f16<EPI_ADD, 2, 4><<<64, 256, 0, stream>>>(
        yacc16, DINNER, wo16, DINNER, ymf, CDIM,
        TOK, CDIM, DINNER, nullptr, xflat, nullptr, 0);

    // 7) LN2 + transpose to (B, C, T, H, W)
    ln2_out_kernel<<<TOK, CDIM, 0, stream>>>(ymf, ln2w, ln2b, out);
}
